// LSTM_85263690760295
// MI455X (gfx1250) — matrix operand order repaired and checked by reading, not compiled, not run on hardware
//
#include <hip/hip_runtime.h>
#include <hip/hip_bf16.h>
#include <stdint.h>

// ---- CDNA5 WMMA vector types (wave32) ----
typedef __attribute__((ext_vector_type(16))) __bf16 v16bf;
typedef __attribute__((ext_vector_type(8)))  __bf16 bf16x8;
typedef __attribute__((ext_vector_type(8)))  float  v8f;

union AFragU { v16bf v; bf16x8 h[2]; };

// ---------------------------------------------------------------------------
// fp32 (K x N, row-major) -> bf16 transposed (Npad x K, row-major), zero pad.
// LDS-tiled 32x32 transpose, 256 threads.
// ---------------------------------------------------------------------------
__global__ __launch_bounds__(256)
void lstm_transpose_f32_bf16(const float* __restrict__ W, __bf16* __restrict__ Wt,
                             int K, int N, int Npad) {
    __shared__ float tile[32][33];
    const int n0 = blockIdx.x * 32;
    const int k0 = blockIdx.y * 32;
    const int tx = threadIdx.x & 31;
    const int ty = threadIdx.x >> 5;   // 0..7
#pragma unroll
    for (int r = 0; r < 32; r += 8) {
        int k = k0 + ty + r, n = n0 + tx;
        tile[ty + r][tx] = (k < K && n < N) ? W[(size_t)k * N + n] : 0.0f;
    }
    __syncthreads();
#pragma unroll
    for (int r = 0; r < 32; r += 8) {
        int n = n0 + ty + r, k = k0 + tx;
        if (n < Npad && k < K) Wt[(size_t)n * K + k] = (__bf16)tile[tx][ty + r];
    }
}

// ---------------------------------------------------------------------------
// Embedding gather: x[bs, e] = bf16(emb[tokens[bs], e])
// ---------------------------------------------------------------------------
__global__ __launch_bounds__(256)
void lstm_embed_gather(const int* __restrict__ tokens, const float* __restrict__ emb,
                       __bf16* __restrict__ x, int BS, int E) {
    size_t i = (size_t)blockIdx.x * blockDim.x + threadIdx.x;
    size_t total = (size_t)BS * E;
    if (i >= total) return;
    int bs = (int)(i / E);
    int e  = (int)(i % E);
    int tok = tokens[bs];
    x[i] = (__bf16)emb[(size_t)tok * E + e];
}

// ---------------------------------------------------------------------------
// Zero-init helpers (workspace is poisoned 0xAA by the harness).
// ---------------------------------------------------------------------------
__global__ __launch_bounds__(256)
void lstm_zero_f32(float* p, int n) {
    int i = blockIdx.x * blockDim.x + threadIdx.x;
    if (i < n) p[i] = 0.0f;
}
__global__ __launch_bounds__(256)
void lstm_zero_bf16(__bf16* p, int n) {
    int i = blockIdx.x * blockDim.x + threadIdx.x;
    if (i < n) p[i] = (__bf16)0.0f;
}

// ---------------------------------------------------------------------------
// Blocked WMMA GEMM: C[M x N] = A[M x K] * Bt[n][k]^T (+ bias[n]).
// One wave computes a 32(M) x 64(N) macro-tile: 2x4 grid of 16x16 WMMA tiles.
// A row-major bf16, Bt is the weight stored transposed [Nrows x K] bf16.
// Bt has >= tilesN4*64 valid (possibly zero-padded) rows; stores/bias are
// guarded against the logical N.
// ---------------------------------------------------------------------------
__global__ __launch_bounds__(256)
void lstm_gemm_bf16_2x4(const __bf16* __restrict__ A, const __bf16* __restrict__ Bt,
                        const float* __restrict__ bias, float* __restrict__ C,
                        int M, int N, int K, int tilesN4) {
    const int wave = blockIdx.x * (blockDim.x >> 5) + (threadIdx.x >> 5);
    const int lane = threadIdx.x & 31;
    const int tM2 = wave / tilesN4;
    const int tN4 = wave % tilesN4;
    const int m0  = tM2 * 32;
    if (m0 >= M) return;

    const int hi   = lane >> 4;     // 0: low half-lanes, 1: high half-lanes
    const int l15  = lane & 15;
    const int asel = hi ? 16 : 0;   // A: elems0-15 = K+asel..
    const int bsel = hi ? 16 : 0;   // B: elems0-15 = K+bsel..

    const __bf16* Arow0 = A + (size_t)(m0 + l15) * K;
    const __bf16* Arow1 = Arow0 + (size_t)16 * K;

    const __bf16* Brow[4];
    int cols[4];
    v8f acc[2][4];
#pragma unroll
    for (int j = 0; j < 4; ++j) {
        int n = (tN4 * 4 + j) * 16 + l15;
        cols[j]  = n;
        Brow[j]  = Bt + (size_t)n * K;
        float bv = (bias != nullptr && n < N) ? bias[n] : 0.0f;
#pragma unroll
        for (int i = 0; i < 2; ++i)
#pragma unroll
            for (int e = 0; e < 8; ++e) acc[i][j][e] = bv;
    }

    for (int k = 0; k < K; k += 32) {
        AFragU a0, a1;
        a0.h[0] = *(const bf16x8*)(Arow0 + k + asel);
        a0.h[1] = *(const bf16x8*)(Arow0 + k + 8 + asel);
        a1.h[0] = *(const bf16x8*)(Arow1 + k + asel);
        a1.h[1] = *(const bf16x8*)(Arow1 + k + 8 + asel);
#pragma unroll
        for (int j = 0; j < 4; ++j) {
            v16bf bfrag = *(const v16bf*)(Brow[j] + k + bsel);
            __builtin_prefetch(Brow[j] + k + 64, 0, 1);   // global_prefetch_b8
            acc[0][j] = __builtin_amdgcn_wmma_f32_16x16x32_bf16(
                false, a0.v, false, bfrag, (short)0, acc[0][j], false, false);
            acc[1][j] = __builtin_amdgcn_wmma_f32_16x16x32_bf16(
                false, a1.v, false, bfrag, (short)0, acc[1][j], false, false);
        }
    }

#pragma unroll
    for (int i = 0; i < 2; ++i) {
        int row0 = m0 + i * 16 + (hi ? 8 : 0);
#pragma unroll
        for (int j = 0; j < 4; ++j) {
            int col = cols[j];
            if (col >= N) continue;
#pragma unroll
            for (int r = 0; r < 8; ++r)
                C[(size_t)(row0 + r) * N + col] = acc[i][j][r];
        }
    }
}

// ---------------------------------------------------------------------------
// Single-tile WMMA GEMM for the tiny recurrent step (M=32): one wave per
// 16x16 tile so the 512 tiles spread across WGPs. No bias, zero-init acc.
// ---------------------------------------------------------------------------
__global__ __launch_bounds__(256)
void lstm_gemm_bf16_1x1(const __bf16* __restrict__ A, const __bf16* __restrict__ Bt,
                        float* __restrict__ C, int M, int N, int K, int tilesN) {
    const int wave = blockIdx.x * (blockDim.x >> 5) + (threadIdx.x >> 5);
    const int lane = threadIdx.x & 31;
    const int tM = wave / tilesN;
    const int tN = wave % tilesN;
    const int m0 = tM * 16;
    if (m0 >= M) return;

    const int hi   = lane >> 4;
    const int l15  = lane & 15;
    const int asel = hi ? 16 : 0;
    const int bsel = hi ? 16 : 0;

    const __bf16* Arow = A + (size_t)(m0 + l15) * K;
    const __bf16* Brow = Bt + (size_t)(tN * 16 + l15) * K;

    v8f acc;
#pragma unroll
    for (int e = 0; e < 8; ++e) acc[e] = 0.0f;

    for (int k = 0; k < K; k += 32) {
        AFragU a;
        a.h[0] = *(const bf16x8*)(Arow + k + asel);
        a.h[1] = *(const bf16x8*)(Arow + k + 8 + asel);
        v16bf bfrag = *(const v16bf*)(Brow + k + bsel);
        acc = __builtin_amdgcn_wmma_f32_16x16x32_bf16(
            false, a.v, false, bfrag, (short)0, acc, false, false);
    }

    int row0 = m0 + (hi ? 8 : 0);
    int col  = tN * 16 + l15;
#pragma unroll
    for (int r = 0; r < 8; ++r)
        C[(size_t)(row0 + r) * N + col] = acc[r];
}

// ---------------------------------------------------------------------------
// Fused LSTM cell (gate order i,f,g,o). z = xz[:,t,:] + zh.
// Updates fp32 c/h state, bf16 h state (next-step GEMM input), and writes h
// into the bf16 sequence buffer hseq[(b*S + t)*H + j].
// ---------------------------------------------------------------------------
__global__ __launch_bounds__(256)
void lstm_cell(const float* __restrict__ xz, const float* __restrict__ zh,
               float* __restrict__ c, float* __restrict__ hf,
               __bf16* __restrict__ hb, __bf16* __restrict__ hseq,
               int t, int S, int H, int B) {
    int idx = blockIdx.x * blockDim.x + threadIdx.x;
    if (idx >= B * H) return;
    int b = idx / H;
    int j = idx % H;
    const int G = 4 * H;
    size_t zoff = ((size_t)b * S + t) * G + j;
    size_t hoff = (size_t)b * G + j;

    float zi = xz[zoff]             + zh[hoff];
    float zf = xz[zoff + H]         + zh[hoff + H];
    float zg = xz[zoff + 2 * (size_t)H] + zh[hoff + 2 * (size_t)H];
    float zo = xz[zoff + 3 * (size_t)H] + zh[hoff + 3 * (size_t)H];

    float ig = 1.0f / (1.0f + __expf(-zi));
    float fg = 1.0f / (1.0f + __expf(-zf));
    float gg = tanhf(zg);
    float og = 1.0f / (1.0f + __expf(-zo));

    float cn = fg * c[idx] + ig * gg;
    float hn = og * tanhf(cn);

    c[idx]  = cn;
    hf[idx] = hn;
    hb[idx] = (__bf16)hn;
    hseq[((size_t)b * S + t) * H + j] = (__bf16)hn;
}

// ---------------------------------------------------------------------------
// Host orchestration. Inputs (fp32/int32) in setup_inputs order:
// tokens, emb, Wi0, Wh0, b0, Wi1, Wh1, b1, Wout, bout.
// d_out = logits (B,S,V) fp32, then c0, h0, c1, h1 (each B*H fp32).
// ---------------------------------------------------------------------------
extern "C" void kernel_launch(void* const* d_in, const int* in_sizes, int n_in,
                              void* d_out, int out_size, void* d_ws, size_t ws_size,
                              hipStream_t stream) {
    (void)in_sizes; (void)n_in; (void)out_size; (void)ws_size;
    constexpr int cB = 32, cS = 512, cV = 10000, cE = 1024, cH = 1024;
    constexpr int G    = 4 * cH;      // 4096
    constexpr int BS   = cB * cS;     // 16384
    constexpr int Vpad = 10048;       // multiple of 64 >= cV (zero-padded rows)

    const int*   tokens = (const int*)d_in[0];
    const float* emb    = (const float*)d_in[1];
    const float* Wi0    = (const float*)d_in[2];
    const float* Wh0    = (const float*)d_in[3];
    const float* b0     = (const float*)d_in[4];
    const float* Wi1    = (const float*)d_in[5];
    const float* Wh1    = (const float*)d_in[6];
    const float* b1     = (const float*)d_in[7];
    const float* Wout   = (const float*)d_in[8];
    const float* bout   = (const float*)d_in[9];
    float* out = (float*)d_out;

    // ---- workspace layout (all 256B aligned), ~391 MB total ----
    char* ws = (char*)d_ws;
    size_t off = 0;
    auto take = [&](size_t bytes) -> char* {
        char* p = ws + off;
        off += (bytes + 255) & ~(size_t)255;
        return p;
    };
    float*  XZ    = (float*) take((size_t)BS * G * 4);    // 268 MB, reused both layers
    __bf16* xbuf  = (__bf16*)take((size_t)BS * cE * 2);   // x bf16; reused as hseq1
    __bf16* hseq0 = (__bf16*)take((size_t)BS * cH * 2);
    __bf16* Wi0t  = (__bf16*)take((size_t)G * cE * 2);
    __bf16* Wh0t  = (__bf16*)take((size_t)G * cH * 2);
    __bf16* Wi1t  = (__bf16*)take((size_t)G * cH * 2);
    __bf16* Wh1t  = (__bf16*)take((size_t)G * cH * 2);
    __bf16* Woutt = (__bf16*)take((size_t)Vpad * cH * 2);
    float*  ZH    = (float*) take((size_t)cB * G * 4);
    float*  c0    = (float*) take((size_t)cB * cH * 4);
    float*  h0f   = (float*) take((size_t)cB * cH * 4);
    __bf16* h0b   = (__bf16*)take((size_t)cB * cH * 2);
    float*  c1    = (float*) take((size_t)cB * cH * 4);
    float*  h1f   = (float*) take((size_t)cB * cH * 4);
    __bf16* h1b   = (__bf16*)take((size_t)cB * cH * 2);
    __bf16* hseq1 = xbuf;

    // ---- 1. weight conversion + transpose to bf16 [N][K] ----
    lstm_transpose_f32_bf16<<<dim3(G / 32, cE / 32), 256, 0, stream>>>(Wi0, Wi0t, cE, G, G);
    lstm_transpose_f32_bf16<<<dim3(G / 32, cH / 32), 256, 0, stream>>>(Wh0, Wh0t, cH, G, G);
    lstm_transpose_f32_bf16<<<dim3(G / 32, cH / 32), 256, 0, stream>>>(Wi1, Wi1t, cH, G, G);
    lstm_transpose_f32_bf16<<<dim3(G / 32, cH / 32), 256, 0, stream>>>(Wh1, Wh1t, cH, G, G);
    lstm_transpose_f32_bf16<<<dim3(Vpad / 32, cH / 32), 256, 0, stream>>>(Wout, Woutt, cH, cV, Vpad);

    // ---- 2. embedding gather -> bf16 activations ----
    lstm_embed_gather<<<((size_t)BS * cE) / 256, 256, 0, stream>>>(tokens, emb, xbuf, BS, cE);

    const int bhBlocks = (cB * cH) / 256;
    const size_t tail = (size_t)BS * cV;   // start of carries in d_out

    // ---- 3. layer 0: xz = x @ Wi0 + b0 (one big WMMA GEMM) ----
    {
        int tilesN4 = G / 64;                       // 64
        int waves   = (BS / 32) * tilesN4;          // 32768
        lstm_gemm_bf16_2x4<<<waves * 32 / 256, 256, 0, stream>>>(
            xbuf, Wi0t, b0, XZ, BS, G, cE, tilesN4);
    }
    lstm_zero_f32<<<bhBlocks, 256, 0, stream>>>(c0, cB * cH);
    lstm_zero_bf16<<<bhBlocks, 256, 0, stream>>>(h0b, cB * cH);
    for (int t = 0; t < cS; ++t) {
        int tilesN = G / 16;                        // 256
        int waves  = (cB / 16) * tilesN;            // 512
        lstm_gemm_bf16_1x1<<<waves * 32 / 256, 256, 0, stream>>>(
            h0b, Wh0t, ZH, cB, G, cH, tilesN);
        lstm_cell<<<bhBlocks, 256, 0, stream>>>(XZ, ZH, c0, h0f, h0b, hseq0, t, cS, cH, cB);
    }
    hipMemcpyAsync(out + tail,                 c0,  (size_t)cB * cH * 4, hipMemcpyDeviceToDevice, stream);
    hipMemcpyAsync(out + tail + (size_t)cB * cH, h0f, (size_t)cB * cH * 4, hipMemcpyDeviceToDevice, stream);

    // ---- 4. layer 1: xz = h0 @ Wi1 + b1 (reuse XZ) ----
    {
        int tilesN4 = G / 64;
        int waves   = (BS / 32) * tilesN4;
        lstm_gemm_bf16_2x4<<<waves * 32 / 256, 256, 0, stream>>>(
            hseq0, Wi1t, b1, XZ, BS, G, cH, tilesN4);
    }
    lstm_zero_f32<<<bhBlocks, 256, 0, stream>>>(c1, cB * cH);
    lstm_zero_bf16<<<bhBlocks, 256, 0, stream>>>(h1b, cB * cH);
    for (int t = 0; t < cS; ++t) {
        int tilesN = G / 16;
        int waves  = (cB / 16) * tilesN;
        lstm_gemm_bf16_1x1<<<waves * 32 / 256, 256, 0, stream>>>(
            h1b, Wh1t, ZH, cB, G, cH, tilesN);
        lstm_cell<<<bhBlocks, 256, 0, stream>>>(XZ, ZH, c1, h1f, h1b, hseq1, t, cS, cH, cB);
    }
    hipMemcpyAsync(out + tail + 2 * (size_t)cB * cH, c1,  (size_t)cB * cH * 4, hipMemcpyDeviceToDevice, stream);
    hipMemcpyAsync(out + tail + 3 * (size_t)cB * cH, h1f, (size_t)cB * cH * 4, hipMemcpyDeviceToDevice, stream);

    // ---- 5. logits = h1 @ Wout + bout ----
    {
        int tilesN4 = Vpad / 64;                    // 157
        int waves   = (BS / 32) * tilesN4;          // 80384
        lstm_gemm_bf16_2x4<<<waves * 32 / 256, 256, 0, stream>>>(
            hseq1, Woutt, bout, out, BS, cV, cH, tilesN4);
    }
}
